// SpikeToGraphFNO2D_11003706212834
// MI455X (gfx1250) — matrix operand order repaired and checked by reading, not compiled, not run on hardware
//
#include <hip/hip_runtime.h>
#include <hip/hip_bf16.h>
#include <math.h>

// ---------------------------------------------------------------------------
// SpikeToGraphFNO2D for MI455X (gfx1250, wave32, WMMA)
// B=2, N=512, T=4096, D=128, MODES=32, EH=256, EH2=128
// ---------------------------------------------------------------------------

typedef __attribute__((ext_vector_type(16))) _Float16 v16h;
typedef __attribute__((ext_vector_type(8)))  float    v8f;
typedef __attribute__((ext_vector_type(2)))  float    v2f;
typedef __attribute__((ext_vector_type(4)))  int      v4i;

#define NB      512
#define TT      4096
#define DM      128
#define NMODES  32
#define EH      256
#define EH2     128
#define ROWS    1024          // B*N
#define NN2     262144        // 512*512
#define PI2     6.28318530717958647692f
#define AST     264           // padded A-tile row stride in halfs (16B aligned, bank-spread)

#ifndef __has_builtin
#define __has_builtin(x) 0
#endif
#if __has_builtin(__builtin_amdgcn_global_load_async_to_lds_b128)
#define HAVE_ASYNC_LDS 1
#else
#define HAVE_ASYNC_LDS 0
#endif

// ---------------- workspace layout (floats) --------------------------------
#define OFF_XR    0u
#define OFF_XI    32768u
#define OFF_E     65536u
#define OFF_T1    196608u
#define OFF_E2    327680u
#define OFF_T2    458752u
#define OFF_H     589824u
#define OFF_HS    720896u
#define OFF_HT    983040u
#define OFF_PS    1245184u
#define OFF_PT    1507328u
#define OFF_CRE   1769472u
#define OFF_CIM   2293760u
#define OFF_TC    2818048u
#define OFF_TS    3080192u
#define OFF_D1R   3342336u
#define OFF_D1I   3866624u
#define OFF_EW2H  4390912u    // 32768 halfs = 16384 floats
// total ~4,407,296 floats = ~16.8 MB

// ---------------------------------------------------------------------------
// K0: twiddle matrices  Tc[j*512+v] = cos(2pi jv/512)/512, Ts = sin(...)/512
// ---------------------------------------------------------------------------
__global__ void twiddle_kernel(float* __restrict__ TC, float* __restrict__ TS) {
    int idx = blockIdx.x * blockDim.x + threadIdx.x;      // 0..262143
    int j = idx >> 9, v = idx & 511;
    float ang = (PI2 / 512.0f) * (float)((j * v) & 511);
    TC[idx] = __cosf(ang) * (1.0f / 512.0f);
    TS[idx] = __sinf(ang) * (1.0f / 512.0f);
}

// ---------------------------------------------------------------------------
// K1: ew2 (f32 [256][128]) -> f16
// ---------------------------------------------------------------------------
__global__ void ew2half_kernel(const float* __restrict__ ew2, _Float16* __restrict__ out) {
    int idx = blockIdx.x * blockDim.x + threadIdx.x;      // 0..32767
    out[idx] = (_Float16)ew2[idx];
}

// ---------------------------------------------------------------------------
// K2: 32-mode rDFT.  One block per (b,n) row; 8 waves, lane = mode,
// wave = 512-sample chunk, phase-rotation recurrence + stream prefetch.
// ---------------------------------------------------------------------------
__global__ void __launch_bounds__(256) dft32_kernel(const float* __restrict__ x,
                                                    float* __restrict__ XR,
                                                    float* __restrict__ XI) {
    int row = blockIdx.x;
    const float* xr = x + (size_t)row * TT;
    int tid = threadIdx.x;
    int w = tid >> 5, k = tid & 31;
    int t0 = w * 512;
    float th0 = (PI2 / (float)TT) * (float)(k * t0);
    float dth = (PI2 / (float)TT) * (float)k;
    float c = __cosf(th0), s = __sinf(th0);
    float cd = __cosf(dth), sd = __sinf(dth);
    float re = 0.0f, im = 0.0f;
    for (int tb = 0; tb < 512; tb += 128) {
        if (tb + 128 < 512) __builtin_prefetch(&xr[t0 + tb + 128], 0, 0);
        for (int t = tb; t < tb + 128; ++t) {
            float xv = xr[t0 + t];
            re = fmaf(xv, c, re);
            im = fmaf(-xv, s, im);
            float cn = c * cd - s * sd;
            s = fmaf(s, cd, c * sd);
            c = cn;
        }
    }
    __shared__ float red[8][32][2];
    red[w][k][0] = re;
    red[w][k][1] = im;
    __syncthreads();
    if (tid < 32) {
        float sr = 0.0f, si = 0.0f;
        for (int ww = 0; ww < 8; ++ww) { sr += red[ww][tid][0]; si += red[ww][tid][1]; }
        XR[row * NMODES + tid] = sr;
        XI[row * NMODES + tid] = si;
    }
}

// ---------------------------------------------------------------------------
// K3: spectral energy  E[row][d] = mean_k |(xr+ixi)*(Wr+iWi)| (+eps)
// ---------------------------------------------------------------------------
__global__ void __launch_bounds__(128) energy_kernel(const float* __restrict__ XR,
                                                     const float* __restrict__ XI,
                                                     const float* __restrict__ Wr,
                                                     const float* __restrict__ Wi,
                                                     float* __restrict__ E) {
    int row = blockIdx.x, d = threadIdx.x;
    const float* xr = XR + row * NMODES;
    const float* xi = XI + row * NMODES;
    float sum = 0.0f;
    for (int k = 0; k < NMODES; ++k) {
        float a = xr[k], bb = xi[k];
        float wr = Wr[k * DM + d], wi = Wi[k * DM + d];
        float yr = a * wr - bb * wi;
        float yi = a * wi + bb * wr;
        sum += sqrtf(yr * yr + yi * yi + 1e-8f);
    }
    E[row * DM + d] = sum * (1.0f / (float)NMODES);
}

// ---------------------------------------------------------------------------
// K4: generic row GEMM: out[row][n] = act(in[row][:K] @ W[K][Nc] + bias[n])
// ---------------------------------------------------------------------------
__global__ void rowgemm_kernel(const float* __restrict__ in, const float* __restrict__ W,
                               const float* __restrict__ bias, float* __restrict__ out,
                               int K, int Nc, int doRelu) {
    int row = blockIdx.x;
    __shared__ float rs[256];
    for (int idx = threadIdx.x; idx < K; idx += blockDim.x) rs[idx] = in[(size_t)row * K + idx];
    __syncthreads();
    int n = threadIdx.x;
    float acc = bias ? bias[n] : 0.0f;
    for (int k = 0; k < K; ++k) acc = fmaf(rs[k], W[k * Nc + n], acc);
    if (doRelu) acc = fmaxf(acc, 0.0f);
    out[(size_t)row * Nc + n] = acc;
}

// ---------------------------------------------------------------------------
// K5: fused edge MLP (34-GFLOP hot spot, f16 WMMA).
// Block = 16 i x 16 j tile = 256 edge rows, 8 waves.
// h is regenerated (never in HBM). Per M-tile the 16x256 f16 A-tile is built
// cooperatively ONCE into a double-buffered LDS buffer (packed cvt + b128
// stores); waves then fetch A fragments as two ds_load_b128 each (the WMMA
// A frag is two 8-half K runs per lane, table order). B fragments (ew2 f16) preloaded once
// per wave in 64 VGPRs. Stage2 (+eb2, relu, x ew3) reduced via LDS f32 atomics.
// ps/pt staging uses the CDNA5 async global->LDS path (ASYNCcnt).
// ---------------------------------------------------------------------------
__global__ void __launch_bounds__(256) edge_kernel(const float* __restrict__ PS,
                                                   const float* __restrict__ PT,
                                                   const float* __restrict__ eb1,
                                                   const _Float16* __restrict__ ew2h,
                                                   const float* __restrict__ eb2,
                                                   const float* __restrict__ ew3,
                                                   const float* __restrict__ eb3,
                                                   float* __restrict__ CRE,
                                                   float* __restrict__ CIM) {
    int bidx = blockIdx.x;               // 0..2047
    int b   = bidx >> 10;
    int rem = bidx & 1023;
    int i0  = (rem >> 5) << 4;           // i tile base
    int j0  = (rem & 31) << 4;           // j tile base

    __shared__ float ps_s[16][EH];       // becomes ps + eb1
    __shared__ float pt_s[16][EH];
    __shared__ float eb1_s[EH];
    __shared__ float rt_s[256 * 2];
    __shared__ alignas(16) _Float16 Ah[2][16 * AST];   // double-buffered f16 A tile

    int tid = threadIdx.x;
    const float* psrc = PS + ((size_t)(b * NB + i0)) * EH;
    const float* ptrc = PT + ((size_t)(b * NB + j0)) * EH;

#if HAVE_ASYNC_LDS
    // async global->LDS staging of ps/pt (16KB each): 4 b128 per thread per array
    {
        for (int c = 0; c < 4; ++c) {
            int q = tid + c * 256;                       // int4 index, 0..1023
            __builtin_amdgcn_global_load_async_to_lds_b128(
                (__attribute__((address_space(1))) v4i*)(psrc + q * 4),
                (__attribute__((address_space(3))) v4i*)(&ps_s[0][0] + q * 4),
                0, 0);
            __builtin_amdgcn_global_load_async_to_lds_b128(
                (__attribute__((address_space(1))) v4i*)(ptrc + q * 4),
                (__attribute__((address_space(3))) v4i*)(&pt_s[0][0] + q * 4),
                0, 0);
        }
#if __has_builtin(__builtin_amdgcn_s_wait_asynccnt)
        __builtin_amdgcn_s_wait_asynccnt(0);
#else
        asm volatile("s_wait_asynccnt 0" ::: "memory");
#endif
    }
#else
    for (int idx = tid; idx < 16 * EH; idx += 256) {
        ps_s[idx >> 8][idx & 255] = psrc[idx];
        pt_s[idx >> 8][idx & 255] = ptrc[idx];
    }
#endif
    eb1_s[tid] = eb1[tid];
    rt_s[tid * 2 + 0] = eb3[0];
    rt_s[tid * 2 + 1] = eb3[1];
    __syncthreads();

    // fold eb1 into ps once (ps_s <- ps + eb1)
    for (int idx = tid; idx < 16 * EH; idx += 256)
        ps_s[idx >> 8][idx & 255] += eb1_s[idx & 255];
    __syncthreads();

    int w = tid >> 5, lane = tid & 31;
    int lm = lane & 15;
    int n  = w * 16 + lm;                        // this wave's output column
    int kOffB  = (lane < 16) ? 0 : 16;           // unused; B frags use kHalfA
    int kHalfA = (lane < 16) ? 0 : 8;            // A frag K offset

    // preload B fragments (whole K=256 chain, 8 steps of 32) -> 64 VGPRs
    v16h bfr[8];
#pragma unroll
    for (int ks = 0; ks < 8; ++ks)
#pragma unroll
        for (int e = 0; e < 16; ++e)
            bfr[ks][e] = ew2h[(ks * 32 + kHalfA + (e < 8 ? e : e + 8)) * EH2 + n];

    float w3a = ew3[n * 2 + 0];
    float w3b = ew3[n * 2 + 1];
    float bb2 = eb2[n];

    // cooperative A-tile build: A[j_loc][k] = relu(ps_s[mt][k] + pt_s[j_loc][k])
    // thread -> (j = tid>>4, 16 contiguous k starting at (tid&15)*16)
    int bj = tid >> 4;
    int bk = (tid & 15) << 4;
    auto buildA = [&](int mt, int buf) {
        const float* pse = &ps_s[mt][0];
        const float* ptj = &pt_s[bj][0];
        union { _Float16 h[16]; uint4 q[2]; } tmpu;
#pragma unroll
        for (int e = 0; e < 16; ++e)
            tmpu.h[e] = (_Float16)fmaxf(pse[bk + e] + ptj[bk + e], 0.0f);
        uint4* dst = (uint4*)&Ah[buf][bj * AST + bk];
        dst[0] = tmpu.q[0];
        dst[1] = tmpu.q[1];
    };

    buildA(0, 0);
    __syncthreads();

    for (int mt = 0; mt < 16; ++mt) {            // M-tile: i_loc = mt, rows = j_loc
        int buf = mt & 1;
        if (mt + 1 < 16) buildA(mt + 1, buf ^ 1);

        v8f acc = {};
        const _Float16* arow = &Ah[buf][lm * AST + kHalfA];
#pragma unroll
        for (int ks = 0; ks < 8; ++ks) {
            union { v16h v; uint4 q[2]; } af;
            const uint4* pa = (const uint4*)(arow + ks * 32);
            af.q[0] = pa[0];                     // K = kb+khalf+0..7
            af.q[1] = pa[2];                     // K = kb+khalf+16..23
            acc = __builtin_amdgcn_wmma_f32_16x16x32_f16(
                false, af.v, false, bfr[ks], (short)0, acc, false, false);
        }
        // stage2: +eb2, relu, contract 128 -> 2 via ew3, reduce across columns
        int rbase = mt * 16 + 8 * (lane >> 4);
#pragma unroll
        for (int r = 0; r < 8; ++r) {
            float v = fmaxf(acc[r] + bb2, 0.0f);
            int er = rbase + r;
            atomicAdd(&rt_s[er * 2 + 0], v * w3a);
            atomicAdd(&rt_s[er * 2 + 1], v * w3b);
        }
        __syncthreads();
    }

    int il = tid >> 4, jl = tid & 15;
    size_t o = ((size_t)(b * NB + i0 + il)) * NB + (j0 + jl);
    CRE[o] = rt_s[tid * 2 + 0];
    CIM[o] = rt_s[tid * 2 + 1];
}

// ---------------------------------------------------------------------------
// K6: ifft along last axis as f32 WMMA GEMM:
//   D1r = Cr@Tc - Ci@Ts ; D1i = Cr@Ts + Ci@Tc   (Tc/Ts already /512)
// ---------------------------------------------------------------------------
__global__ void __launch_bounds__(256) idft_pass1(const float* __restrict__ CRE,
                                                  const float* __restrict__ CIM,
                                                  const float* __restrict__ TC,
                                                  const float* __restrict__ TS,
                                                  float* __restrict__ D1R,
                                                  float* __restrict__ D1I) {
    int w = threadIdx.x >> 5, lane = threadIdx.x & 31;
    int g   = blockIdx.x * 8 + w;                // 0..2047 wave tiles
    int b   = g >> 10;
    int rem = g & 1023;
    int mr  = (rem >> 5) << 4;
    int nc  = (rem & 31) << 4;
    int lm  = lane & 15;
    int kA  = (lane < 16) ? 0 : 2;

    const float* Cr = CRE + (size_t)b * NN2;
    const float* Ci = CIM + (size_t)b * NN2;

    v8f aRR = {}, aIS = {}, aRS = {}, aIC = {};
    for (int kk = 0; kk < 512; kk += 4) {
        int arow = (mr + lm) * 512 + kk + kA;
        v2f ar = *(const v2f*)(&Cr[arow]);
        v2f ai = *(const v2f*)(&Ci[arow]);
        v2f bc, bs;
        int brow = (kk + kA) * 512 + nc + lm;
        bc[0] = TC[brow]; bc[1] = TC[brow + 512];
        bs[0] = TS[brow]; bs[1] = TS[brow + 512];
        aRR = __builtin_amdgcn_wmma_f32_16x16x4_f32(false, ar, false, bc, (short)0, aRR, false, false);
        aIS = __builtin_amdgcn_wmma_f32_16x16x4_f32(false, ai, false, bs, (short)0, aIS, false, false);
        aRS = __builtin_amdgcn_wmma_f32_16x16x4_f32(false, ar, false, bs, (short)0, aRS, false, false);
        aIC = __builtin_amdgcn_wmma_f32_16x16x4_f32(false, ai, false, bc, (short)0, aIC, false, false);
    }
#pragma unroll
    for (int r = 0; r < 8; ++r) {
        int row = mr + r + 8 * (lane >> 4);
        size_t o = ((size_t)(b * NB + row)) * NB + nc + lm;
        D1R[o] = aRR[r] - aIS[r];
        D1I[o] = aRS[r] + aIC[r];
    }
}

// ---------------------------------------------------------------------------
// K7: ifft along first axis + affine + sigmoid.
// ---------------------------------------------------------------------------
__global__ void __launch_bounds__(256) idft_pass2(const float* __restrict__ D1R,
                                                  const float* __restrict__ D1I,
                                                  const float* __restrict__ TC,
                                                  const float* __restrict__ TS,
                                                  const float* __restrict__ alpha_p,
                                                  const float* __restrict__ beta_p,
                                                  float* __restrict__ out) {
    int w = threadIdx.x >> 5, lane = threadIdx.x & 31;
    int g   = blockIdx.x * 8 + w;
    int b   = g >> 10;
    int rem = g & 1023;
    int mr  = (rem >> 5) << 4;
    int nc  = (rem & 31) << 4;
    int lm  = lane & 15;
    int kA  = (lane < 16) ? 0 : 2;
    float alpha = alpha_p[0], beta = beta_p[0];

    v8f a1 = {}, a2 = {};
    for (int kk = 0; kk < 512; kk += 4) {
        int arow = (mr + lm) * 512 + kk + kA;
        v2f atc = *(const v2f*)(&TC[arow]);
        v2f ats = *(const v2f*)(&TS[arow]);
        v2f brv, biv;
        size_t brow = ((size_t)(b * NB + kk + kA)) * NB + nc + lm;
        brv[0] = D1R[brow]; brv[1] = D1R[brow + 512];
        biv[0] = D1I[brow]; biv[1] = D1I[brow + 512];
        a1 = __builtin_amdgcn_wmma_f32_16x16x4_f32(false, atc, false, brv, (short)0, a1, false, false);
        a2 = __builtin_amdgcn_wmma_f32_16x16x4_f32(false, ats, false, biv, (short)0, a2, false, false);
    }
#pragma unroll
    for (int r = 0; r < 8; ++r) {
        int row = mr + r + 8 * (lane >> 4);
        size_t o = ((size_t)(b * NB + row)) * NB + nc + lm;
        float logit = alpha * (a1[r] - a2[r]) + beta;
        out[o] = logit;
        out[(size_t)2 * NN2 + o] = 1.0f / (1.0f + __expf(-logit));
    }
}

// ---------------------------------------------------------------------------
extern "C" void kernel_launch(void* const* d_in, const int* in_sizes, int n_in,
                              void* d_out, int out_size, void* d_ws, size_t ws_size,
                              hipStream_t stream) {
    (void)in_sizes; (void)n_in; (void)out_size; (void)ws_size;
    const float* spikes = (const float*)d_in[0];
    const float* Wr  = (const float*)d_in[1];
    const float* Wi  = (const float*)d_in[2];
    const float* pw1 = (const float*)d_in[3];
    const float* pb1 = (const float*)d_in[4];
    const float* pw2 = (const float*)d_in[5];
    const float* pb2 = (const float*)d_in[6];
    const float* nw1 = (const float*)d_in[7];
    const float* nb1 = (const float*)d_in[8];
    const float* nw2 = (const float*)d_in[9];
    const float* nb2 = (const float*)d_in[10];
    const float* sw  = (const float*)d_in[11];
    const float* sb  = (const float*)d_in[12];
    const float* tw  = (const float*)d_in[13];
    const float* tb  = (const float*)d_in[14];
    const float* ew1 = (const float*)d_in[15];
    const float* eb1 = (const float*)d_in[16];
    const float* ew2 = (const float*)d_in[17];
    const float* eb2 = (const float*)d_in[18];
    const float* ew3 = (const float*)d_in[19];
    const float* eb3 = (const float*)d_in[20];
    const float* alpha = (const float*)d_in[21];
    const float* beta  = (const float*)d_in[22];

    float* ws = (float*)d_ws;
    float* XR  = ws + OFF_XR;   float* XI  = ws + OFF_XI;
    float* E   = ws + OFF_E;    float* T1  = ws + OFF_T1;
    float* E2  = ws + OFF_E2;   float* T2  = ws + OFF_T2;
    float* H   = ws + OFF_H;
    float* HS  = ws + OFF_HS;   float* HT  = ws + OFF_HT;
    float* PS  = ws + OFF_PS;   float* PT  = ws + OFF_PT;
    float* CRE = ws + OFF_CRE;  float* CIM = ws + OFF_CIM;
    float* TC  = ws + OFF_TC;   float* TS  = ws + OFF_TS;
    float* D1R = ws + OFF_D1R;  float* D1I = ws + OFF_D1I;
    _Float16* EW2H = (_Float16*)(ws + OFF_EW2H);
    float* out = (float*)d_out;

    // constant prep
    twiddle_kernel<<<NN2 / 256, 256, 0, stream>>>(TC, TS);
    ew2half_kernel<<<(EH * EH2) / 256, 256, 0, stream>>>(ew2, EW2H);

    // temporal FNO front-end
    dft32_kernel<<<ROWS, 256, 0, stream>>>(spikes, XR, XI);
    energy_kernel<<<ROWS, 128, 0, stream>>>(XR, XI, Wr, Wi, E);

    // row MLP chain (1024 rows, tiny)
    rowgemm_kernel<<<ROWS, 128, 0, stream>>>(E,  pw1, pb1, T1, 128, 128, 1);
    rowgemm_kernel<<<ROWS, 128, 0, stream>>>(T1, pw2, pb2, E2, 128, 128, 0);
    rowgemm_kernel<<<ROWS, 128, 0, stream>>>(E2, nw1, nb1, T2, 128, 128, 1);
    rowgemm_kernel<<<ROWS, 128, 0, stream>>>(T2, nw2, nb2, H,  128, 128, 1);
    rowgemm_kernel<<<ROWS, 256, 0, stream>>>(H,  sw,  sb,  HS, 128, 256, 0);
    rowgemm_kernel<<<ROWS, 256, 0, stream>>>(H,  tw,  tb,  HT, 128, 256, 0);
    rowgemm_kernel<<<ROWS, 256, 0, stream>>>(HS, ew1,            nullptr, PS, 256, 256, 0);
    rowgemm_kernel<<<ROWS, 256, 0, stream>>>(HT, ew1 + EH * EH,  nullptr, PT, 256, 256, 0);

    // fused edge MLP (f16 WMMA) -> complex Fourier coeffs
    edge_kernel<<<2048, 256, 0, stream>>>(PS, PT, eb1, EW2H, eb2, ew3, eb3, CRE, CIM);

    // ifft2 as two f32-WMMA DFT passes + affine + sigmoid
    idft_pass1<<<256, 256, 0, stream>>>(CRE, CIM, TC, TS, D1R, D1I);
    idft_pass2<<<256, 256, 0, stream>>>(D1R, D1I, TC, TS, alpha, beta, out);
}
